// DisARM_19481971655249
// MI455X (gfx1250) — matrix operand order repaired and checked by reading, not compiled, not run on hardware
//
#include <hip/hip_runtime.h>
#include <cmath>

typedef __attribute__((ext_vector_type(16))) _Float16 v16h;
typedef __attribute__((ext_vector_type(8)))  _Float16 v8h;
typedef __attribute__((ext_vector_type(8)))  float    v8f;

#define PLANE    65536      // NUM*NA = 1024*64
#define P_TOTAL  524288     // BZ*NUM*NA
#define LSTR     80         // LDS row stride in halfs (160 B, 32B-aligned rows)

// Tile table in d_ws (f16, pre-swizzled into the per-lane WMMA A layout):
//   tiles  0..15 : Wf1 (mt 0..3, kb 0..3)   tile = mt*4 + kb
//   tiles 16..19 : Wf2 (mt 0..1, kb 0..1)   tile = 16 + mt*2 + kb
//   tiles 20..23 : Wa1 (mt 0..1, kb 0..1)   tile = 20 + mt*2 + kb
//   tiles 24..25 : Ws3 (mt 0..1, K zero-padded 16->32)
// Each tile: 32 lanes x 16 halfs = 1024 B; lane's 16 halfs contiguous (32 B).
#define NTILES 26

__device__ __forceinline__ float fast_tanh(float x) {
#if __has_builtin(__builtin_amdgcn_tanhf)
  return __builtin_amdgcn_tanhf(x);          // v_tanh_f32 (CDNA5 TRANS op)
#else
  // tanh(x) = 1 - 2/(exp(2x)+1): v_exp_f32 + ~4 VALU ops, exact formula.
  // x -> +inf: e=inf -> 1 ; x -> -inf: e=0 -> -1 ; NaN propagates.
  const float e = __expf(2.0f * x);
  return 1.0f - 2.0f / (e + 1.0f);
#endif
}

// --------- prepack: f32 weights -> f16 tiles in this kernel's WMMA A lane layout (contiguous k, same as B) ---------
//   lanes 0-15:  M=lane,    K=kb*32+2v,2v+1   (v=0..7)
//   lanes 16-31: M=lane-16, K=kb*32+16+2v,..  (v=0..7)
__global__ __launch_bounds__(32) void pack_weights(
    const float* __restrict__ Wf1, const float* __restrict__ Wf2,
    const float* __restrict__ Wa1, const float* __restrict__ Ws3,
    _Float16* __restrict__ wpk)
{
  const int t    = blockIdx.x;     // tile id
  const int lane = threadIdx.x;    // 0..31
  const float* W; int K, mt, kb; bool padK = false;
  if (t < 16)      { W = Wf1; K = 128; mt = t >> 2;        kb = t & 3;        }
  else if (t < 20) { W = Wf2; K = 64;  mt = (t - 16) >> 1; kb = (t - 16) & 1; }
  else if (t < 24) { W = Wa1; K = 64;  mt = (t - 20) >> 1; kb = (t - 20) & 1; }
  else             { W = Ws3; K = 16;  mt = t - 24;        kb = 0; padK = true; }

  const int m     = mt * 16 + (lane & 15);
  const int kbase = kb * 32 + (lane >> 4) * 16;
  v16h A;
#pragma unroll
  for (int v = 0; v < 8; ++v) {
    const int k0 = kbase + 2 * v;
    float x0 = 0.0f, x1 = 0.0f;
    if (!(padK && k0 >= 16)) { x0 = W[m * K + k0]; x1 = W[m * K + k0 + 1]; }
    A[2 * v]     = (_Float16)x0;
    A[2 * v + 1] = (_Float16)x1;
  }
  *(v16h*)(wpk + (size_t)t * 512 + lane * 16) = A;
}

__global__ __launch_bounds__(128) void disarm_fused(
    const float* __restrict__ loc,  const float* __restrict__ feat,
    const float* __restrict__ bf1,  const float* __restrict__ gf1,
    const float* __restrict__ btf1, const float* __restrict__ bf2,
    const float* __restrict__ Ws1,  const float* __restrict__ bs1,
    const float* __restrict__ gs1,  const float* __restrict__ bts1,
    const float* __restrict__ Ws2,  const float* __restrict__ bs2,
    const float* __restrict__ gs2,  const float* __restrict__ bts2,
    const float* __restrict__ bs3,  const float* __restrict__ ba1,
    const float* __restrict__ ga1,  const float* __restrict__ bta1,
    const float* __restrict__ Wa2,  const float* __restrict__ ba2,
    const _Float16* __restrict__ wpk,
    float* __restrict__ out)
{
  __shared__ __align__(32) _Float16 HA[64][LSTR];  // hidden h  [pixel][64ch]
  __shared__ __align__(32) _Float16 HB[64][LSTR];  // cat tile  [pixel][spa0..31|fd32..63]
  __shared__ float sc1[64], sh1[64];               // BN1 folded scale/shift
  __shared__ float sca[32], sha[32];               // BN-agg folded scale/shift
  __shared__ float sb3[32], sf2[32], sw2[32];      // bs3, bf2, Wa2 caches
  __shared__ float Tl[64], Wl[64];

  const int tid  = threadIdx.x;
  const int lane = tid & 31;
  const int wave = tid >> 5;
  const int n    = lane & 15;      // WMMA B/D column (pixel within wave tile)
  const int half = lane >> 4;      // lane-half selects K/M sub-range
  const int px   = wave * 16 + n;  // pixel (anchor) within workgroup

  const int wg = blockIdx.x;       // (b, num) pair
  const int b  = wg >> 10;
  const int q0 = (wg & 1023) * 64; // pixel-in-plane base
  const float rs = rsqrtf(1.0f + 1e-5f);

  if (tid < 64) { float s = gf1[tid] * rs; sc1[tid] = s; sh1[tid] = bf1[tid] * s + btf1[tid]; }
  if (tid < 32) {
    float s = ga1[tid] * rs; sca[tid] = s; sha[tid] = ba1[tid] * s + bta1[tid];
    sb3[tid] = bs3[tid]; sf2[tid] = bf2[tid]; sw2[tid] = Wa2[tid];
  }
  __syncthreads();

  // ---------------- Phase 1: features 128 -> 64 (WMMA f16) ----------------
  const float* fb = feat + (size_t)b * (128ull * PLANE) + (size_t)(q0 + wave * 16 + n);
  v8f acc[4] = {};
#pragma unroll
  for (int kb = 0; kb < 4; ++kb) {
    if (kb < 3) __builtin_prefetch(fb + (size_t)(kb * 32 + 32) * PLANE, 0, 0);
    // B-tile 32x16: lanes 0-15 hold K = kb*32 + e, lanes 16-31 K = kb*32+16+e
    v16h B;
    const int kB = kb * 32 + half * 16;
#pragma unroll
    for (int e = 0; e < 16; ++e)
      B[e] = (_Float16)fb[(size_t)(kB + e) * PLANE];
#pragma unroll
    for (int mt = 0; mt < 4; ++mt) {
      const v16h A = *(const v16h*)(wpk + (size_t)(mt * 4 + kb) * 512 + lane * 16);
      acc[mt] = __builtin_amdgcn_wmma_f32_16x16x32_f16(
          false, A, false, B, (short)0, acc[mt], false, false);
    }
  }

  // ---------------- Phase 2: BN + tanh, stash h as f16 [px][ch] ----------------
#pragma unroll
  for (int mt = 0; mt < 4; ++mt) {
    v8h h8;
#pragma unroll
    for (int r = 0; r < 8; ++r) {
      const int m = mt * 16 + half * 8 + r;
      h8[r] = (_Float16)fast_tanh(acc[mt][r] * sc1[m] + sh1[m]);
    }
    *(v8h*)(&HA[px][mt * 16 + half * 8]) = h8;   // 8 consecutive channels
  }

  // ---------------- Phase 3: feat_dis 64 -> 32 (WMMA), B from LDS ----------------
  v8f acc2[2] = {};
#pragma unroll
  for (int kb = 0; kb < 2; ++kb) {
    const v16h B = *(const v16h*)(&HA[px][kb * 32 + half * 16]);  // contiguous 16 ch
#pragma unroll
    for (int mt = 0; mt < 2; ++mt) {
      const v16h A = *(const v16h*)(wpk + (size_t)(16 + mt * 2 + kb) * 512 + lane * 16);
      acc2[mt] = __builtin_amdgcn_wmma_f32_16x16x32_f16(
          false, A, false, B, (short)0, acc2[mt], false, false);
    }
  }
#pragma unroll
  for (int mt = 0; mt < 2; ++mt) {
    v8h f8;
#pragma unroll
    for (int r = 0; r < 8; ++r) {
      const int m = mt * 16 + half * 8 + r;
      f8[r] = (_Float16)(acc2[mt][r] + sf2[m]);
    }
    *(v8h*)(&HB[px][32 + mt * 16 + half * 8]) = f8;  // cat: fd in channels 32..63
  }

  // ------------- Phase 4: spatial MLP 3 -> 8 -> 16 (VALU), 16 -> 32 (WMMA) -------------
  {
    const float* lp = loc + ((size_t)b * PLANE + (size_t)(q0 + wave * 16 + n)) * 3;
    const float l0 = lp[0], l1 = lp[1], l2 = lp[2];
    float s1v[8];
#pragma unroll
    for (int o = 0; o < 8; ++o) {
      const float a = Ws1[o * 3 + 0] * l0 + Ws1[o * 3 + 1] * l1 + Ws1[o * 3 + 2] * l2 + bs1[o];
      s1v[o] = fast_tanh(a * (gs1[o] * rs) + bts1[o]);
    }
    float s2v[16];
#pragma unroll
    for (int o = 0; o < 16; ++o) {
      float a = bs2[o];
#pragma unroll
      for (int i = 0; i < 8; ++i) a += Ws2[o * 8 + i] * s1v[i];
      s2v[o] = fast_tanh(a * (gs2[o] * rs) + bts2[o]);
    }
    // 16->32 as WMMA with K padded to 32: lane n holds s2 of its own pixel as
    // B column n (K=0..15 from lanes 0-15); K=16..31 (lanes 16-31) are zeros,
    // matching the zero-padded Ws3 tiles.
    v16h Bs;
#pragma unroll
    for (int e = 0; e < 16; ++e)
      Bs[e] = half ? (_Float16)0.0f : (_Float16)s2v[e];
    v8f accs[2] = {};
#pragma unroll
    for (int mt = 0; mt < 2; ++mt) {
      const v16h A = *(const v16h*)(wpk + (size_t)(24 + mt) * 512 + lane * 16);
      accs[mt] = __builtin_amdgcn_wmma_f32_16x16x32_f16(
          false, A, false, Bs, (short)0, accs[mt], false, false);
    }
#pragma unroll
    for (int mt = 0; mt < 2; ++mt) {
      v8h p8;
#pragma unroll
      for (int r = 0; r < 8; ++r) {
        const int m = mt * 16 + half * 8 + r;
        p8[r] = (_Float16)(accs[mt][r] + sb3[m]);
      }
      *(v8h*)(&HB[px][mt * 16 + half * 8]) = p8;   // cat: spa in channels 0..31
    }
  }

  // ---------------- Phase 5: agg 64 -> 32 (WMMA) + tanh + 32 -> 1 dot ----------------
  v8f acc3[2] = {};
#pragma unroll
  for (int kb = 0; kb < 2; ++kb) {
    const v16h B = *(const v16h*)(&HB[px][kb * 32 + half * 16]);
#pragma unroll
    for (int mt = 0; mt < 2; ++mt) {
      const v16h A = *(const v16h*)(wpk + (size_t)(20 + mt * 2 + kb) * 512 + lane * 16);
      acc3[mt] = __builtin_amdgcn_wmma_f32_16x16x32_f16(
          false, A, false, B, (short)0, acc3[mt], false, false);
    }
  }
  float part = 0.0f;
#pragma unroll
  for (int mt = 0; mt < 2; ++mt) {
#pragma unroll
    for (int r = 0; r < 8; ++r) {
      const int m = mt * 16 + half * 8 + r;
      part += sw2[m] * fast_tanh(acc3[mt][r] * sca[m] + sha[m]);
    }
  }
  part += __shfl_xor(part, 16);               // combine the two lane-halves
  const float t = fast_tanh(part + ba2[0]);   // tanh(out) per anchor
  if (half == 0) Tl[px] = t;
  __syncthreads();

  // ---------------- Phase 6: softmax + min-max over the 64 anchors ----------------
  const float v0 = Tl[lane], v1 = Tl[lane + 32];
  float mx = fmaxf(v0, v1);
#pragma unroll
  for (int off = 16; off > 0; off >>= 1) mx = fmaxf(mx, __shfl_xor(mx, off));
  float se = __expf(v0 - mx) + __expf(v1 - mx);
#pragma unroll
  for (int off = 16; off > 0; off >>= 1) se += __shfl_xor(se, off);

  const float w = __expf(t - mx) / se;
  if (half == 0) Wl[px] = w;
  __syncthreads();

  const float w0 = Wl[lane], w1 = Wl[lane + 32];
  float wmn = fminf(w0, w1), wmx = fmaxf(w0, w1);
#pragma unroll
  for (int off = 16; off > 0; off >>= 1) {
    wmn = fminf(wmn, __shfl_xor(wmn, off));
    wmx = fmaxf(wmx, __shfl_xor(wmx, off));
  }
  const float kk = (1.0f + 1e-6f) / (wmx - wmn + 1e-6f);

  if (half == 0) {
    const size_t p = (size_t)wg * 64 + px;
    out[p]                   = w;
    out[(size_t)P_TOTAL + p] = kk * (w - wmn);
  }
}

extern "C" void kernel_launch(void* const* d_in, const int* in_sizes, int n_in,
                              void* d_out, int out_size, void* d_ws, size_t ws_size,
                              hipStream_t stream) {
  (void)in_sizes; (void)n_in; (void)out_size; (void)ws_size;
  _Float16* wpk = (_Float16*)d_ws;   // 26 KiB of pre-swizzled f16 weight tiles

  pack_weights<<<NTILES, 32, 0, stream>>>(
      (const float*)d_in[2],   // Wf1
      (const float*)d_in[6],   // Wf2
      (const float*)d_in[18],  // Wa1
      (const float*)d_in[16],  // Ws3
      wpk);

  disarm_fused<<<8192, 128, 0, stream>>>(
      (const float*)d_in[0],  (const float*)d_in[1],
      (const float*)d_in[3],  (const float*)d_in[4],  (const float*)d_in[5],
      (const float*)d_in[7],
      (const float*)d_in[8],  (const float*)d_in[9],  (const float*)d_in[10], (const float*)d_in[11],
      (const float*)d_in[12], (const float*)d_in[13], (const float*)d_in[14], (const float*)d_in[15],
      (const float*)d_in[17],
      (const float*)d_in[19], (const float*)d_in[20], (const float*)d_in[21],
      (const float*)d_in[22], (const float*)d_in[23],
      wpk,
      (float*)d_out);
}
